// PointNetSetAbstractionMsg_23270132810092
// MI455X (gfx1250) — matrix operand order repaired and checked by reading, not compiled, not run on hardware
//
#include <hip/hip_runtime.h>

typedef __attribute__((ext_vector_type(16))) _Float16 v16h;
typedef __attribute__((ext_vector_type(8)))  float    v8f;

#define NB   16
#define NPTS 4096
#define NS   1024
#define CIN  16
#define COUT_TOT 323   // 3 + 64 + 128 + 128

// ---------------------------------------------------------------------------
// Farthest point sampling: one block per batch. xyz + running min-dist live in
// LDS (48KB + 16KB). 1024 sequential steps, block-wide argmax with
// first-index tie-break (matches jnp.argmax). Records `far` BEFORE the update
// (matches the lax.scan carry semantics), so sample 0 is always index 0.
// Writes new_xyz to workspace and to d_out channels [0..3).
// ---------------------------------------------------------------------------
__global__ __launch_bounds__(512) void fps_kernel(const float* __restrict__ xyz,
                                                  float* __restrict__ newxyz,
                                                  float* __restrict__ out) {
  __shared__ float sx[NPTS * 3];
  __shared__ float sd[NPTS];
  __shared__ float rv[512];
  __shared__ int   ri[512];
  __shared__ int   sFar;

  const int b = blockIdx.x, tid = threadIdx.x;
  const float* xb = xyz + (size_t)b * NPTS * 3;
  for (int j = tid; j < NPTS * 3; j += 512) sx[j] = xb[j];
  for (int j = tid; j < NPTS; j += 512) sd[j] = 1e10f;
  if (tid == 0) sFar = 0;
  __syncthreads();

  for (int i = 0; i < NS; ++i) {
    const int far = sFar;
    const float cx = sx[far * 3 + 0], cy = sx[far * 3 + 1], cz = sx[far * 3 + 2];
    if (tid == 0) {
      const size_t o = (size_t)b * NS + i;
      newxyz[o * 3 + 0] = cx; newxyz[o * 3 + 1] = cy; newxyz[o * 3 + 2] = cz;
      out[o * COUT_TOT + 0] = cx; out[o * COUT_TOT + 1] = cy; out[o * COUT_TOT + 2] = cz;
    }
    float bv = -1.0f; int bi = 0;
#pragma unroll
    for (int j = 0; j < 8; ++j) {
      const int p = tid * 8 + j;                // contiguous chunk -> ascending idx
      const float dx = sx[p * 3 + 0] - cx;
      const float dy = sx[p * 3 + 1] - cy;
      const float dz = sx[p * 3 + 2] - cz;
      const float d  = dx * dx + dy * dy + dz * dz;
      const float nd = fminf(sd[p], d);
      sd[p] = nd;
      if (nd > bv) { bv = nd; bi = p; }         // strict > keeps first index
    }
    rv[tid] = bv; ri[tid] = bi;
    __syncthreads();
    for (int stp = 256; stp > 0; stp >>= 1) {
      if (tid < stp) {
        const float ov = rv[tid + stp]; const int oi = ri[tid + stp];
        if (ov > rv[tid] || (ov == rv[tid] && oi < ri[tid])) { rv[tid] = ov; ri[tid] = oi; }
      }
      __syncthreads();
    }
    if (tid == 0) sFar = ri[0];
    __syncthreads();
  }
}

// ---------------------------------------------------------------------------
// One branch of the MSG layer. One wave32 per centroid, 4 waves per block.
//   1) ball query: ballot + prefix popcount selects the first KS indices in
//      ascending order (== reference's where/sort/slice), pads with argmin.
//   2) gathered 19-ch input built directly into the WMMA A-operand register
//      layout (16x32 f16): element e of lane L holds k=16*(e>>3)+8*(L>>4)+(e&7)
//      for row m=L&15 (zero padded k=19..31).
//   3) 3 dense layers via v_wmma_f32_16x16x32_f16, weights staged in LDS as
//      f16 [k][n] so a B operand (32x16) is lane L = column n=L&15, 16 k rows
//      read with row stride, same k order as A. Inter-layer C->A relayout via a per-wave LDS tile.
//   4) gamma/beta/ReLU per output column (N = lane&15), running max over the
//      KS/16 row tiles, cross-half shfl_xor(16), store.
// ---------------------------------------------------------------------------
template <int KS, int C1, int C2, int C3, int OFF>
__global__ __launch_bounds__(128) void msg_branch(
    const float* __restrict__ xyz, const float* __restrict__ feat,
    const float* __restrict__ newxyz,
    const float* __restrict__ w1, const float* __restrict__ g1, const float* __restrict__ b1,
    const float* __restrict__ w2, const float* __restrict__ g2, const float* __restrict__ b2,
    const float* __restrict__ w3, const float* __restrict__ g3, const float* __restrict__ b3,
    float* __restrict__ out, float radius) {
  constexpr int CACT = (C1 > C2) ? C1 : C2;
  __shared__ _Float16 sW1[32 * C1];
  __shared__ _Float16 sW2[C1 * C2];
  __shared__ _Float16 sW3[C2 * C3];
  __shared__ float sg1[C1], sb1[C1], sg2[C2], sb2[C2], sg3[C3], sb3[C3];
  __shared__ int      sIdx[4][KS];
  __shared__ _Float16 sAct[4][16 * CACT];

  const int tid = threadIdx.x;
  // ---- stage weights (f32 -> f16), layer1 zero-padded 19 -> 32 rows ----
  for (int i = tid; i < 32 * C1; i += 128) {
    const int k = i / C1, c = i - k * C1;
    sW1[i] = (_Float16)((k < 19) ? w1[k * C1 + c] : 0.0f);
  }
  for (int i = tid; i < C1 * C2; i += 128) sW2[i] = (_Float16)w2[i];
  for (int i = tid; i < C2 * C3; i += 128) sW3[i] = (_Float16)w3[i];
  for (int i = tid; i < C1; i += 128) { sg1[i] = g1[i]; sb1[i] = b1[i]; }
  for (int i = tid; i < C2; i += 128) { sg2[i] = g2[i]; sb2[i] = b2[i]; }
  for (int i = tid; i < C3; i += 128) { sg3[i] = g3[i]; sb3[i] = b3[i]; }

  const int lane = tid & 31, w = tid >> 5;
  const int m = lane & 15, h = lane >> 4;
  const int gid = blockIdx.x * 4 + w;
  const int b = gid >> 10, s = gid & 1023;
  const size_t cbase = (size_t)b * NS + s;
  const float cx = newxyz[cbase * 3 + 0];
  const float cy = newxyz[cbase * 3 + 1];
  const float cz = newxyz[cbase * 3 + 2];
  const float* xb = xyz  + (size_t)b * NPTS * 3;
  const float* fb = feat + (size_t)b * NPTS * CIN;

  // ---- ball query: first KS indices (ascending) within radius ----
  const float r2 = radius * radius;
  float mnv = 3.4e38f; int mni = 0; unsigned cnt = 0;
  for (int it = 0; it < NPTS / 32; ++it) {
    const int p = it * 32 + lane;
    const float dx = xb[p * 3 + 0] - cx;
    const float dy = xb[p * 3 + 1] - cy;
    const float dz = xb[p * 3 + 2] - cz;
    const float d  = dx * dx + dy * dy + dz * dz;
    const bool pred = (d <= r2);
    const unsigned msk = (unsigned)__ballot(pred);
    if (pred) {
      const unsigned slot = cnt + (unsigned)__popc(msk & ((1u << lane) - 1u));
      if (slot < (unsigned)KS) sIdx[w][slot] = p;
    }
    cnt += (unsigned)__popc(msk);
    if (d < mnv) { mnv = d; mni = p; }
  }
#pragma unroll
  for (int off = 16; off > 0; off >>= 1) {     // wave argmin, lower index wins ties
    const float ov = __shfl_xor(mnv, off, 32);
    const int   oi = __shfl_xor(mni, off, 32);
    if (ov < mnv || (ov == mnv && oi < mni)) { mnv = ov; mni = oi; }
  }
  const int filled = (cnt < (unsigned)KS) ? (int)cnt : KS;
  for (int sl = filled + lane; sl < KS; sl += 32) sIdx[w][sl] = mni;
  __syncthreads();

  // ---- fused 3-layer MLP over KS/16 row tiles, max-pool in registers ----
  float pool[C3 / 16][8];
#pragma unroll
  for (int nt = 0; nt < C3 / 16; ++nt)
#pragma unroll
    for (int i = 0; i < 8; ++i) pool[nt][i] = -3.4e38f;

  for (int t = 0; t < KS / 16; ++t) {
    const int p = sIdx[w][t * 16 + m];
    const float x0 = xb[p * 3 + 0] - cx;
    const float x1 = xb[p * 3 + 1] - cy;
    const float x2 = xb[p * 3 + 2] - cz;
    const float* fr = fb + (size_t)p * CIN;
    v16h a1;
#pragma unroll
    for (int e = 0; e < 16; ++e) {
      const int k = ((e >> 3) << 4) + (h << 3) + (e & 7);
      const float v = (k == 0) ? x0 : (k == 1) ? x1 : (k == 2) ? x2
                      : ((k < 19) ? fr[k - 3] : 0.0f);
      a1[e] = (_Float16)v;
    }
    // ---- layer 1: 19(->32) -> C1 ----
#pragma unroll
    for (int nt = 0; nt < C1 / 16; ++nt) {
      v16h bop;
#pragma unroll
      for (int e = 0; e < 16; ++e) bop[e] = sW1[(((e >> 3) << 4) + (h << 3) + (e & 7)) * C1 + nt * 16 + m];
      v8f c = {0.f, 0.f, 0.f, 0.f, 0.f, 0.f, 0.f, 0.f};
      c = __builtin_amdgcn_wmma_f32_16x16x32_f16(false, a1, false, bop, (short)0, c, false, false);
      const float gg = sg1[nt * 16 + m], bb = sb1[nt * 16 + m];
#pragma unroll
      for (int i = 0; i < 8; ++i) {
        const float f = fmaxf(c[i] * gg + bb, 0.0f);
        sAct[w][(i + 8 * h) * CACT + nt * 16 + m] = (_Float16)f;
      }
    }
    __syncthreads();
    // ---- layer 2: C1 -> C2 ----
    v16h a2[C1 / 32];
#pragma unroll
    for (int kc = 0; kc < C1 / 32; ++kc)
#pragma unroll
      for (int e = 0; e < 16; ++e)
        a2[kc][e] = sAct[w][m * CACT + kc * 32 + ((e >> 3) << 4) + (h << 3) + (e & 7)];
#pragma unroll
    for (int nt = 0; nt < C2 / 16; ++nt) {
      v8f c = {0.f, 0.f, 0.f, 0.f, 0.f, 0.f, 0.f, 0.f};
#pragma unroll
      for (int kc = 0; kc < C1 / 32; ++kc) {
        v16h bop;
#pragma unroll
        for (int e = 0; e < 16; ++e) bop[e] = sW2[(kc * 32 + ((e >> 3) << 4) + (h << 3) + (e & 7)) * C2 + nt * 16 + m];
        c = __builtin_amdgcn_wmma_f32_16x16x32_f16(false, a2[kc], false, bop, (short)0, c, false, false);
      }
      const float gg = sg2[nt * 16 + m], bb = sb2[nt * 16 + m];
#pragma unroll
      for (int i = 0; i < 8; ++i) {
        const float f = fmaxf(c[i] * gg + bb, 0.0f);
        sAct[w][(i + 8 * h) * CACT + nt * 16 + m] = (_Float16)f;
      }
    }
    __syncthreads();
    // ---- layer 3: C2 -> C3, fold into running max ----
    v16h a3[C2 / 32];
#pragma unroll
    for (int kc = 0; kc < C2 / 32; ++kc)
#pragma unroll
      for (int e = 0; e < 16; ++e)
        a3[kc][e] = sAct[w][m * CACT + kc * 32 + ((e >> 3) << 4) + (h << 3) + (e & 7)];
#pragma unroll
    for (int nt = 0; nt < C3 / 16; ++nt) {
      v8f c = {0.f, 0.f, 0.f, 0.f, 0.f, 0.f, 0.f, 0.f};
#pragma unroll
      for (int kc = 0; kc < C2 / 32; ++kc) {
        v16h bop;
#pragma unroll
        for (int e = 0; e < 16; ++e) bop[e] = sW3[(kc * 32 + ((e >> 3) << 4) + (h << 3) + (e & 7)) * C3 + nt * 16 + m];
        c = __builtin_amdgcn_wmma_f32_16x16x32_f16(false, a3[kc], false, bop, (short)0, c, false, false);
      }
      const float gg = sg3[nt * 16 + m], bb = sb3[nt * 16 + m];
#pragma unroll
      for (int i = 0; i < 8; ++i)
        pool[nt][i] = fmaxf(pool[nt][i], fmaxf(c[i] * gg + bb, 0.0f));
    }
    __syncthreads();
  }

  // ---- reduce over rows (M in VGPR index + lane half), store ----
#pragma unroll
  for (int nt = 0; nt < C3 / 16; ++nt) {
    float mx = pool[nt][0];
#pragma unroll
    for (int i = 1; i < 8; ++i) mx = fmaxf(mx, pool[nt][i]);
    const float o = __shfl_xor(mx, 16, 32);
    mx = fmaxf(mx, o);
    if (h == 0) out[cbase * COUT_TOT + OFF + nt * 16 + m] = mx;
  }
}

// ---------------------------------------------------------------------------
extern "C" void kernel_launch(void* const* d_in, const int* in_sizes, int n_in,
                              void* d_out, int out_size, void* d_ws, size_t ws_size,
                              hipStream_t stream) {
  (void)in_sizes; (void)n_in; (void)out_size; (void)ws_size;
  const float* xyz  = (const float*)d_in[0];
  const float* feat = (const float*)d_in[1];
  float* out = (float*)d_out;
  float* nx  = (float*)d_ws;   // new_xyz scratch: 16*1024*3 floats (192KB)

  fps_kernel<<<NB, 512, 0, stream>>>(xyz, nx, out);

  const int nblk = NB * NS / 4;  // one wave per centroid, 4 waves per block
#define PW(i) ((const float*)d_in[i])
  msg_branch<16, 32, 32, 64, 3><<<nblk, 128, 0, stream>>>(
      xyz, feat, nx, PW(2), PW(3), PW(4), PW(5), PW(6), PW(7), PW(8), PW(9), PW(10),
      out, 0.1f);
  msg_branch<32, 64, 64, 128, 67><<<nblk, 128, 0, stream>>>(
      xyz, feat, nx, PW(11), PW(12), PW(13), PW(14), PW(15), PW(16), PW(17), PW(18), PW(19),
      out, 0.2f);
  msg_branch<64, 64, 96, 128, 195><<<nblk, 128, 0, stream>>>(
      xyz, feat, nx, PW(20), PW(21), PW(22), PW(23), PW(24), PW(25), PW(26), PW(27), PW(28),
      out, 0.4f);
#undef PW
}
